// Generator_4733053960224
// MI455X (gfx1250) — matrix operand order repaired and checked by reading, not compiled, not run on hardware
//
#include <hip/hip_runtime.h>
#include <hip/hip_bf16.h>
#include <cstdint>
#include <cstddef>

// ---------------------------------------------------------------------------
// CDNA5 (gfx1250) GRU decoder: embed -> x-gates GEMM -> 64-step GRU -> vocab
// projection. GEMMs use v_wmma_f32_16x16x32_bf16 with a double-buffered LDS
// pipeline fed by GLOBAL_LOAD_ASYNC_TO_LDS (ASYNCcnt tracked).
// ---------------------------------------------------------------------------

typedef __attribute__((ext_vector_type(16))) __bf16 v16bf;
typedef __attribute__((ext_vector_type(8)))  float  v8f;

#if defined(__HIP_DEVICE_COMPILE__) && \
    __has_builtin(__builtin_amdgcn_global_load_async_to_lds_b128)
#define USE_ASYNC_LDS 1
#else
#define USE_ASYNC_LDS 0
#endif

static __device__ __forceinline__ unsigned short f32_to_bf16_bits(float f) {
  unsigned u = __float_as_uint(f);
  u += 0x7FFFu + ((u >> 16) & 1u);   // round-to-nearest-even
  return (unsigned short)(u >> 16);
}

// 16B global -> LDS: async DMA path (no VGPR staging) or uint4 fallback.
static __device__ __forceinline__ void copy16B_to_lds(const unsigned short* g,
                                                      unsigned short* l) {
#if USE_ASYNC_LDS
  typedef int v4i __attribute__((vector_size(16)));
  typedef __attribute__((address_space(1))) v4i* gv4i_p;
  typedef __attribute__((address_space(3))) v4i* lv4i_p;
  gv4i_p gp = (gv4i_p)(uintptr_t)g;
  lv4i_p lp = (lv4i_p)(unsigned)(uintptr_t)l;
  __builtin_amdgcn_global_load_async_to_lds_b128(gp, lp, 0, 0);
#else
  *reinterpret_cast<uint4*>(l) = *reinterpret_cast<const uint4*>(g);
#endif
}

static __device__ __forceinline__ void wait_stage_ready() {
#if USE_ASYNC_LDS
#if __has_builtin(__builtin_amdgcn_s_wait_asynccnt)
  __builtin_amdgcn_s_wait_asynccnt(0);
#else
  asm volatile("s_wait_asynccnt 0x0" ::: "memory");
#endif
#endif
}

// ---- weight cast (fp32 -> bf16 bits) with K padding to multiple of 32 -----
__global__ void cast_pad_bf16_kernel(const float* __restrict__ src,
                                     unsigned short* __restrict__ dst,
                                     int rows, int K, int Kpad) {
  size_t i = (size_t)blockIdx.x * blockDim.x + threadIdx.x;
  size_t total = (size_t)rows * Kpad;
  if (i >= total) return;
  int r = (int)(i / Kpad);
  int k = (int)(i % Kpad);
  float f = (k < K) ? src[(size_t)r * K + k] : 0.0f;
  dst[i] = f32_to_bf16_bits(f);
}

// ---- embedding gather -> bf16 A-matrix [B*S, Epad] ------------------------
__global__ void gather_embed_kernel(const int* __restrict__ tgt,
                                    const float* __restrict__ emb,
                                    unsigned short* __restrict__ dst,
                                    int BS, int E, int Epad) {
  size_t i = (size_t)blockIdx.x * blockDim.x + threadIdx.x;
  size_t total = (size_t)BS * Epad;
  if (i >= total) return;
  int r = (int)(i / Epad);
  int e = (int)(i % Epad);
  float f = 0.0f;
  if (e < E) f = emb[(size_t)tgt[r] * E + e];
  dst[i] = f32_to_bf16_bits(f);
}

// ---- h0 = sentence_vec (fp32 state + bf16 copy for WMMA) ------------------
__global__ void init_h_kernel(const float* __restrict__ sv,
                              float* __restrict__ h,
                              unsigned short* __restrict__ hbf, int n) {
  int i = blockIdx.x * blockDim.x + threadIdx.x;
  if (i >= n) return;
  float f = sv[i];
  h[i] = f;
  hbf[i] = f32_to_bf16_bits(f);
}

// ---- fused GRU gates: r,z,n + state update + bf16 emission ----------------
__global__ void gru_gate_kernel(const float* __restrict__ xg,   // [B*S, 3H]
                                const float* __restrict__ hg,   // [B, 3H] (incl b_hh)
                                float* __restrict__ h,          // [B, H] in/out
                                unsigned short* __restrict__ hbf,   // [B, H]
                                unsigned short* __restrict__ hsbf,  // [B*S, H]
                                int t) {
  const int S = 64, H = 2400, B = 32;
  int i = blockIdx.x * blockDim.x + threadIdx.x;
  if (i >= B * H) return;
  int b = i / H, j = i % H;
  size_t xrow = (size_t)(b * S + t) * (3 * H);
  size_t hrow = (size_t)b * (3 * H);
  float xr = xg[xrow + j], xz = xg[xrow + H + j], xn = xg[xrow + 2 * H + j];
  float hr = hg[hrow + j], hz = hg[hrow + H + j], hn = hg[hrow + 2 * H + j];
  float r = 1.0f / (1.0f + __expf(-(xr + hr)));
  float z = 1.0f / (1.0f + __expf(-(xz + hz)));
  float n = tanhf(xn + r * hn);
  float hnew = (1.0f - z) * n + z * h[i];
  h[i] = hnew;
  unsigned short hb = f32_to_bf16_bits(hnew);
  hbf[i] = hb;                               // next recurrence step A-operand
  hsbf[(size_t)(b * S + t) * H + j] = hb;    // row of output-projection A
}

// ---------------------------------------------------------------------------
// Double-buffered LDS-tiled bf16 WMMA GEMM: C[M,N] = A[M,K]*B[N,K]^T + bias.
// 256 threads = 8 waves. Each wave computes a WMxWN register tile:
// (WM/16) A-frags x (WN/16) B-frags -> (WM/16)*(WN/16) WMMA per K-step.
// Bounds handled by clamping row indices; surplus staging threads wrap onto
// low chunks (duplicate 16B loads, identical data) -> the whole inner loop is
// branch-free and EXEC stays all-ones (WMMA requirement).
// K must be a multiple of 32 (buffers are padded).
// ---------------------------------------------------------------------------
template <int BM, int BN, int WM, int WN>
__global__ __launch_bounds__(256) void gemm_bf16_wmma(
    const unsigned short* __restrict__ A,   // [M,K] bf16 bits
    const unsigned short* __restrict__ Bm,  // [N,K] bf16 bits
    const float* __restrict__ bias,         // [N]
    float* __restrict__ C,                  // [M,N] fp32
    int M, int N, int K) {
  constexpr int BK   = 32;
  constexpr int LDSS = BK + 8;              // ushort stride; rows stay 16B-aligned
  constexpr int WAVES_M = BM / WM;
  constexpr int WAVES_N = BN / WN;
  static_assert(WAVES_M * WAVES_N == 8, "need exactly 8 wave32 per block");
  constexpr int MI = WM / 16;               // A frags per wave
  constexpr int NI = WN / 16;               // B frags per wave
  constexpr int A_TOT = BM * BK / 8;        // 16B chunks per stage (pow2)
  constexpr int B_TOT = BN * BK / 8;
  static_assert((A_TOT & (A_TOT - 1)) == 0 && (B_TOT & (B_TOT - 1)) == 0,
                "chunk counts must be powers of two for wrap-mapping");
  constexpr int A_IT  = (A_TOT + 255) / 256;
  constexpr int B_IT  = (B_TOT + 255) / 256;

  __shared__ unsigned short As[2][BM * LDSS];
  __shared__ unsigned short Bs[2][BN * LDSS];

  const int tid  = threadIdx.x;
  const int lane = tid & 31;
  const int wave = tid >> 5;
  const int wm = wave / WAVES_N;
  const int wn = wave % WAVES_N;
  const int mBase = blockIdx.y * BM;
  const int nBase = blockIdx.x * BN;
  const int row16 = lane & 15;
  const int half  = lane >> 4;              // 0/1: K/M half held by this lane

  // -- precompute per-thread staging addresses (clamped/wrapped, no branches)
  const unsigned short* aSrc[A_IT];
  int aLds[A_IT];
#pragma unroll
  for (int c = 0; c < A_IT; ++c) {
    int lc = (c * 256 + tid) & (A_TOT - 1); // wrap: surplus threads duplicate
    int r  = lc >> 2;                       // BK/8 = 4 chunks per row
    int kc = (lc & 3) * 8;
    int gr = mBase + r; if (gr > M - 1) gr = M - 1;
    aSrc[c] = A + (size_t)gr * K + kc;
    aLds[c] = r * LDSS + kc;
  }
  const unsigned short* bSrc[B_IT];
  int bLds[B_IT];
#pragma unroll
  for (int c = 0; c < B_IT; ++c) {
    int lc = (c * 256 + tid) & (B_TOT - 1);
    int r  = lc >> 2;
    int kc = (lc & 3) * 8;
    int gr = nBase + r; if (gr > N - 1) gr = N - 1;
    bSrc[c] = Bm + (size_t)gr * K + kc;
    bLds[c] = r * LDSS + kc;
  }

  auto stage = [&](int buf, int k0) {
#pragma unroll
    for (int c = 0; c < A_IT; ++c)
      copy16B_to_lds(aSrc[c] + k0, &As[buf][aLds[c]]);
#pragma unroll
    for (int c = 0; c < B_IT; ++c)
      copy16B_to_lds(bSrc[c] + k0, &Bs[buf][bLds[c]]);
  };

  v8f acc[MI][NI];
#pragma unroll
  for (int mi = 0; mi < MI; ++mi)
#pragma unroll
    for (int ni = 0; ni < NI; ++ni)
#pragma unroll
      for (int i = 0; i < 8; ++i) acc[mi][ni][i] = 0.0f;

  union Frag { v16bf v; unsigned short s[16]; };

  const int nsteps = K / BK;
  stage(0, 0);

  for (int i = 0; i < nsteps; ++i) {
    const int cur = i & 1;
    wait_stage_ready();                     // s_wait_asynccnt 0
    __syncthreads();                        // stage `cur` visible to all waves
    if (i + 1 < nsteps) stage(cur ^ 1, (i + 1) * BK);  // overlap next stage

    // ---- fragments per documented CDNA5 VGPR layouts ---------------------
    // A (16x32 MxK): lanes 0-15 hold K {0..7,16..23}, lanes 16-31 {8..15,24..31}
    Frag af[MI];
#pragma unroll
    for (int mi = 0; mi < MI; ++mi) {
      const unsigned short* ap = &As[cur][(wm * WM + mi * 16 + row16) * LDSS];
#pragma unroll
      for (int v = 0; v < 8; ++v) {
        int kb = ((v < 4) ? 2 * v : 16 + 2 * (v - 4)) + 8 * half;
        af[mi].s[2 * v]     = ap[kb];
        af[mi].s[2 * v + 1] = ap[kb + 1];
      }
    }
    // B (32x16 KxN): lane holds column N=lane&15, K {0..7,16..23} / {8..15,24..31} by lane>>4
    Frag bf[NI];
    const int kbase = half * 8;
#pragma unroll
    for (int ni = 0; ni < NI; ++ni) {
      const unsigned short* bp = &Bs[cur][(wn * WN + ni * 16 + row16) * LDSS + kbase];
#pragma unroll
      for (int v = 0; v < 8; ++v) {
        int kb = (v < 4) ? 2 * v : 16 + 2 * (v - 4);
        bf[ni].s[2 * v]     = bp[kb];
        bf[ni].s[2 * v + 1] = bp[kb + 1];
      }
    }
    // ---- MI x NI independent accumulator chains -> deep XDL pipelining ---
#pragma unroll
    for (int mi = 0; mi < MI; ++mi)
#pragma unroll
      for (int ni = 0; ni < NI; ++ni)
        acc[mi][ni] = __builtin_amdgcn_wmma_f32_16x16x32_bf16(
            false, af[mi].v, false, bf[ni].v, (short)0, acc[mi][ni], false, false);
    __syncthreads();                        // protect `cur` until all waves done
  }

  // ---- store C: D-layout M = v + 8*(lane>>4), N = lane&15 -----------------
#pragma unroll
  for (int ni = 0; ni < NI; ++ni) {
    int c = nBase + wn * WN + ni * 16 + row16;
    if (c >= N) continue;
    float bv = bias[c];
#pragma unroll
    for (int mi = 0; mi < MI; ++mi) {
#pragma unroll
      for (int v = 0; v < 8; ++v) {
        int r = mBase + wm * WM + mi * 16 + half * 8 + v;
        if (r < M) C[(size_t)r * N + c] = acc[mi][ni][v] + bv;
      }
    }
  }
}

// ---------------------------------------------------------------------------
extern "C" void kernel_launch(void* const* d_in, const int* in_sizes, int n_in,
                              void* d_out, int out_size, void* d_ws, size_t ws_size,
                              hipStream_t stream) {
  (void)in_sizes; (void)n_in; (void)out_size; (void)ws_size;
  const int V = 20000, E = 300, H = 2400, B = 32, S = 64;
  const int BS = B * S;          // 2048
  const int G  = 3 * H;          // 7200
  const int Ep = 320;            // E padded to multiple of 32

  const int*   tgt   = (const int*)  d_in[0];
  const float* sv    = (const float*)d_in[1];
  const float* emb   = (const float*)d_in[2];
  const float* W_ih  = (const float*)d_in[3];
  const float* W_hh  = (const float*)d_in[4];
  const float* b_ih  = (const float*)d_in[5];
  const float* b_hh  = (const float*)d_in[6];
  const float* W_out = (const float*)d_in[7];
  const float* b_out = (const float*)d_in[8];
  float* logits = (float*)d_out;                  // [B,S,V] fp32

  // carve workspace (~207 MB total)
  char* p = (char*)d_ws;
  auto take = [&](size_t bytes) -> char* {
    char* r = p;
    p += (bytes + 255) & ~(size_t)255;
    return r;
  };
  unsigned short* Wih_bf  = (unsigned short*)take((size_t)G  * Ep * 2);
  unsigned short* Whh_bf  = (unsigned short*)take((size_t)G  * H  * 2);
  unsigned short* Wout_bf = (unsigned short*)take((size_t)V  * H  * 2);
  unsigned short* A1_bf   = (unsigned short*)take((size_t)BS * Ep * 2);
  unsigned short* hbf     = (unsigned short*)take((size_t)B  * H  * 2);
  unsigned short* hsbf    = (unsigned short*)take((size_t)BS * H  * 2);
  float* xgates = (float*)take((size_t)BS * G * 4);
  float* hgbuf  = (float*)take((size_t)B  * G * 4);
  float* hbuf   = (float*)take((size_t)B  * H * 4);

  // ---- one-shot prep: bf16 weight casts, embedding gather, h0 ------------
  {
    size_t n;
    n = (size_t)G * Ep;
    cast_pad_bf16_kernel<<<(unsigned)((n + 255) / 256), 256, 0, stream>>>(W_ih, Wih_bf, G, E, Ep);
    n = (size_t)G * H;
    cast_pad_bf16_kernel<<<(unsigned)((n + 255) / 256), 256, 0, stream>>>(W_hh, Whh_bf, G, H, H);
    n = (size_t)V * H;
    cast_pad_bf16_kernel<<<(unsigned)((n + 255) / 256), 256, 0, stream>>>(W_out, Wout_bf, V, H, H);
    n = (size_t)BS * Ep;
    gather_embed_kernel<<<(unsigned)((n + 255) / 256), 256, 0, stream>>>(tgt, emb, A1_bf, BS, E, Ep);
    n = (size_t)B * H;
    init_h_kernel<<<(unsigned)((n + 255) / 256), 256, 0, stream>>>(sv, hbuf, hbf, (int)n);
  }

  // ---- GEMM 1: x_gates[2048,7200] = emb * W_ih^T + b_ih -------------------
  {
    dim3 grid((G + 127) / 128, (BS + 127) / 128);
    gemm_bf16_wmma<128, 128, 32, 64><<<grid, 256, 0, stream>>>(
        A1_bf, Wih_bf, b_ih, xgates, BS, G, Ep);
  }

  // ---- GRU recurrence: 64 sequential (GEMM + fused-gates) steps -----------
  // W_hh (34.6 MB bf16) stays resident in the 192 MB L2 across steps.
  for (int t = 0; t < S; ++t) {
    dim3 grid((G + 255) / 256, 1);
    gemm_bf16_wmma<32, 256, 16, 64><<<grid, 256, 0, stream>>>(
        hbf, Whh_bf, b_hh, hgbuf, B, G, H);
    int n = B * H;
    gru_gate_kernel<<<(n + 255) / 256, 256, 0, stream>>>(xgates, hgbuf, hbuf, hbf, hsbf, t);
  }

  // ---- GEMM 3: logits[2048,20000] = hs * W_out^T + b_out (dominant ~197 GF)
  {
    dim3 grid((V + 127) / 128, (BS + 127) / 128);
    gemm_bf16_wmma<128, 128, 32, 64><<<grid, 256, 0, stream>>>(
        hsbf, Wout_bf, b_out, logits, BS, V, H);
  }
}
